// GravNet_ragged_68599217652088
// MI455X (gfx1250) — compile-verified, matrix operand order checked by reading, not run on hardware
//
#include <hip/hip_runtime.h>

typedef __attribute__((ext_vector_type(16))) _Float16 v16h;
typedef __attribute__((ext_vector_type(8)))  float    v8f;

#define B_   16
#define N_   2048
#define F_   64
#define S_   4
#define P_   22
#define O_   42
#define M_TOT (B_ * N_)   // 32768
#define KNBR  39
#define KPAD  128         // 108 -> 128 (final GEMM K)
#define NPAD  48          // 42  -> 48  (final GEMM N)
#define NCAT  32          // 26  -> 32  (coords+feats N)

// ---- workspace byte offsets (total ~11.3 MB) ----
#define WS_COORDS 0u          // 32768*4  f32 =  524288
#define WS_FEATS  524288u     // 32768*22 f32 = 2883584
#define WS_UPD    3407872u    // 32768*128 f16= 8388608
#define WS_WCT    11796480u   // 32*64  f16   =    4096
#define WS_BCAT   11800576u   // 32 f32       =     128
#define WS_WOT    11800704u   // 48*128 f16   =   12288
#define WS_BO     11812992u   // 48 f32       =     192

union HFrag { v16h v; _Float16 h[16]; unsigned u[8]; };

// ---------------------------------------------------------------------------
// Tensor Data Mover: 1-D copy of n f16 elements from global -> LDS.
// D# built per CDNA5 ISA §8.3/§8.4:
//   group0: count=1 | lds_addr[63:32] | global_addr[95:64],[120:96] | type=2
//   group1: data_size=1(2B), tensor_dim0=n, tensor_dim1=1, tile_dim0=n,
//           tile_dim1/2=0 (unused), tensor_dim0_stride=n
//   groups 2/3 (+ extra group of the 6-arg builtin): zero (<=2-D tensor)
// ---------------------------------------------------------------------------
__device__ inline void tdm_load_1d_f16(unsigned lds_off, const void* gsrc, unsigned n) {
  typedef __attribute__((ext_vector_type(4))) unsigned u32x4;
  typedef __attribute__((ext_vector_type(8))) int      i32x8;
  typedef __attribute__((ext_vector_type(4))) int      i32x4;
  const unsigned long long ga = (unsigned long long)(uintptr_t)gsrc;
  u32x4 g0 = { 1u,                                   // count=1 (valid), user mode
               lds_off,                              // lds_addr
               (unsigned)ga,                         // global_addr[31:0]
               (unsigned)(ga >> 32) | (2u << 30) };  // global_addr[56:32] | type=2
  i32x8 g1 = { (int)(1u << 16),                      // data_size=1 (2 bytes)
               (int)((n & 0xFFFFu) << 16),           // tensor_dim0[15:0]
               (int)((n >> 16) | (1u << 16)),        // tensor_dim0[31:16] | tensor_dim1=1
               (int)((n & 0xFFFFu) << 16),           // tile_dim0 = n
               0,                                    // tile_dim1=0, tile_dim2=0 (unused)
               (int)n,                               // tensor_dim0_stride[31:0]
               0, 0 };
  i32x4 z4 = { 0, 0, 0, 0 };
  i32x8 z8 = { 0, 0, 0, 0, 0, 0, 0, 0 };
  __builtin_amdgcn_tensor_load_to_lds(g0, g1, z4, z4, z8, 0);
}

// Load a 16x32 16-bit A/B fragment from an f32 row-major source (row stride ld).
// Layout (ISA 7.12.2): lanes 0-15 -> row=lane, halves[0..7]=K0..7, halves[8..15]=K16..23
//                      lanes16-31 -> row=lane-16, K8..15 and K24..31.
__device__ inline v16h load_frag_f32(const float* __restrict__ base, int ld, int k0) {
  const int lane = threadIdx.x & 31;
  const int row  = lane & 15;
  const int kh   = (lane >> 4) * 8;
  const float* p = base + (size_t)row * ld + k0 + kh;
  HFrag f;
#pragma unroll
  for (int g = 0; g < 8; ++g) f.h[g] = (_Float16)p[g];
  const float* p2 = p + 16;
#pragma unroll
  for (int g = 0; g < 8; ++g) f.h[8 + g] = (_Float16)p2[g];
  return f.v;
}

// Same fragment from an f16 row-major source (packed 2x16b dword loads).
// Works for global or LDS pointers (clang infers the address space).
__device__ inline v16h load_frag_f16(const _Float16* base, int ld, int k0) {
  const int lane = threadIdx.x & 31;
  const int row  = lane & 15;
  const int kh   = (lane >> 4) * 8;
  const _Float16* p = base + (size_t)row * ld + k0 + kh;
  HFrag f;
  const unsigned* pu = (const unsigned*)p;
#pragma unroll
  for (int g = 0; g < 4; ++g) f.u[g] = pu[g];
  const unsigned* pu2 = (const unsigned*)(p + 16);
#pragma unroll
  for (int g = 0; g < 4; ++g) f.u[4 + g] = pu2[g];
  return f.v;
}

// ---------------- weight repack: W_s||W_f -> WcT[n][k] f16, W_o -> WoT[n][k] f16 ----------
__global__ void k_prep(const float* __restrict__ Ws, const float* __restrict__ bs,
                       const float* __restrict__ Wf, const float* __restrict__ bf,
                       const float* __restrict__ Wo, const float* __restrict__ bo_in,
                       _Float16* __restrict__ WcT, float* __restrict__ bcat,
                       _Float16* __restrict__ WoT, float* __restrict__ bo) {
  const int tid = blockIdx.x * blockDim.x + threadIdx.x;
  if (tid < NCAT * F_) {                        // 2048: WcT (transposed, n-major)
    const int n = tid & (NCAT - 1);
    const int k = tid >> 5;
    float v = 0.f;
    if (n < S_)            v = Ws[k * S_ + n];
    else if (n < S_ + P_)  v = Wf[k * P_ + (n - S_)];
    WcT[n * F_ + k] = (_Float16)v;
  } else if (tid < NCAT * F_ + NCAT) {          // bias_cat
    const int n = tid - NCAT * F_;
    bcat[n] = (n < S_) ? bs[n] : ((n < S_ + P_) ? bf[n - S_] : 0.f);
  } else if (tid < NCAT * F_ + NCAT + NPAD * KPAD) {   // 6144: WoT (transposed, padded)
    const int t = tid - (NCAT * F_ + NCAT);
    const int k = t & (KPAD - 1);
    const int n = t >> 7;
    const float v = (n < O_ && k < (F_ + 2 * P_)) ? Wo[k * O_ + n] : 0.f;
    WoT[n * KPAD + k] = (_Float16)v;
  } else if (tid < NCAT * F_ + NCAT + NPAD * KPAD + NPAD) {
    const int n = tid - (NCAT * F_ + NCAT + NPAD * KPAD);
    bo[n] = (n < O_) ? bo_in[n] : 0.f;
  }
}

// ---------------- kernel 1: coords/feats GEMM via WMMA; also seed updated[:,0:64] ----------
__global__ __launch_bounds__(256) void k_gemm1(const float* __restrict__ x,
        const _Float16* __restrict__ WcT, const float* __restrict__ bcat,
        float* __restrict__ coords, float* __restrict__ feats,
        _Float16* __restrict__ upd) {
  __shared__ _Float16 sWc[NCAT * F_];           // 4 KB: B tile shared by all 8 waves
  const int lane = threadIdx.x & 31;
  const int wave = threadIdx.x >> 5;
  const int tile = blockIdx.x * 8 + wave;       // 2048 tiles total
  const int m0 = tile * 16;

  if (wave == 0) {                              // TDM: WcT -> LDS, once per workgroup
    tdm_load_1d_f16((unsigned)(uintptr_t)(void*)sWc, WcT, NCAT * F_);
    __builtin_amdgcn_s_wait_tensorcnt(0);
  }
  __syncthreads();

  __builtin_prefetch(x + (size_t)(m0 + 128) * F_, 0, 0);   // global_prefetch_b8

  const v16h a0 = load_frag_f32(x + (size_t)m0 * F_, F_, 0);
  const v16h a1 = load_frag_f32(x + (size_t)m0 * F_, F_, 32);

#pragma unroll
  for (int nt = 0; nt < 2; ++nt) {
    const v16h b0 = load_frag_f16(sWc + nt * 16 * F_, F_, 0);
    const v16h b1 = load_frag_f16(sWc + nt * 16 * F_, F_, 32);
    v8f acc = {};
    acc = __builtin_amdgcn_wmma_f32_16x16x32_f16(false, a0, false, b0, (short)0, acc, false, false);
    acc = __builtin_amdgcn_wmma_f32_16x16x32_f16(false, a1, false, b1, (short)0, acc, false, false);

    const int n  = lane & 15;
    const int mb = (lane >> 4) * 8;
    const int col = nt * 16 + n;
    const float bias = bcat[col];
#pragma unroll
    for (int r = 0; r < 8; ++r) {
      const int m = m0 + mb + r;
      const float vv = acc[r] + bias;
      if (col < S_)             coords[(size_t)m * S_ + col] = vv;
      else if (col < S_ + P_)   feats[(size_t)m * P_ + (col - S_)] = vv;
    }
  }

  // updated[:, 0:64] = x (f16); zero the K pad columns 108..127
  for (int idx = lane; idx < 16 * F_; idx += 32) {
    const int r = idx >> 6, c = idx & 63;
    upd[(size_t)(m0 + r) * KPAD + c] = (_Float16)x[(size_t)(m0 + r) * F_ + c];
  }
  for (int idx = lane; idx < 16 * (KPAD - 108); idx += 32) {
    const int r = idx / 20, c = 108 + idx % 20;
    upd[(size_t)(m0 + r) * KPAD + c] = (_Float16)0.f;
  }
}

// ---------------- kernel 2: per-point kNN (39 nearest, self excluded) + weighted max/mean ---
__global__ __launch_bounds__(256) void k_knn(const float* __restrict__ coords,
        const float* __restrict__ feats, _Float16* __restrict__ upd) {
  __shared__ float dls[8][N_];        // 64 KB: one distance row per wave
  const int lane = threadIdx.x & 31;
  const int wave = threadIdx.x >> 5;
  const int row  = blockIdx.x * 8 + wave;       // 0..32767
  const int b = row >> 11;
  const int i = row & (N_ - 1);
  float* dl = dls[wave];

  const float c0 = coords[(size_t)row * 4 + 0];
  const float c1 = coords[(size_t)row * 4 + 1];
  const float c2 = coords[(size_t)row * 4 + 2];
  const float c3 = coords[(size_t)row * 4 + 3];

  const float* cb = coords + (size_t)b * N_ * 4;
  for (int j = lane; j < N_; j += 32) {
    const float d0 = cb[j * 4 + 0] - c0;
    const float d1 = cb[j * 4 + 1] - c1;
    const float d2 = cb[j * 4 + 2] - c2;
    const float d3 = cb[j * 4 + 3] - c3;
    const float d = d0 * d0 + d1 * d1 + d2 * d2 + d3 * d3;
    dl[j] = (j == i) ? __builtin_inff() : d;    // exclude self
  }

  // per-lane running min over its strided 64-candidate subset
  float mv = __builtin_inff(); int mi = 0;
  for (int j = lane; j < N_; j += 32)
    if (dl[j] < mv) { mv = dl[j]; mi = j; }

  float nmax = -__builtin_inff();
  float nsum = 0.f;
  const float* fb = feats + (size_t)b * N_ * P_;

  for (int t = 0; t < KNBR; ++t) {
    // wave32 butterfly argmin -> all lanes converge on (v, id)
    float v = mv; int id = mi;
#pragma unroll
    for (int off = 16; off > 0; off >>= 1) {
      const float ov = __shfl_xor(v, off, 32);
      const int   oi = __shfl_xor(id, off, 32);
      if (ov < v || (ov == v && oi < id)) { v = ov; id = oi; }
    }
    // owner lane removes the winner and rescans its subset
    if ((id & 31) == lane) {
      dl[id] = __builtin_inff();
      mv = __builtin_inff(); mi = 0;
      for (int j = lane; j < N_; j += 32)
        if (dl[j] < mv) { mv = dl[j]; mi = j; }
    }
    // aggregate: lanes 0..21 each own one feature channel
    const float w = expf(-v);
    if (lane < P_) {
      const float fv = fb[(size_t)id * P_ + lane] * w;
      nmax = fmaxf(nmax, fv);
      nsum += fv;
    }
  }

  if (lane < P_) {
    upd[(size_t)row * KPAD + 64 + lane]      = (_Float16)nmax;
    upd[(size_t)row * KPAD + 64 + P_ + lane] = (_Float16)(nsum * (1.0f / KNBR));
  }
}

// ---------------- kernel 3: final GEMM (32768 x 128 x 48) via WMMA + bias + tanh ------------
__global__ __launch_bounds__(256) void k_gemm2(const _Float16* __restrict__ upd,
        const _Float16* __restrict__ WoT, const float* __restrict__ bo,
        float* __restrict__ out) {
  __shared__ _Float16 sWo[NPAD * KPAD];         // 12 KB: B tile shared by all 8 waves
  const int lane = threadIdx.x & 31;
  const int wave = threadIdx.x >> 5;
  const int tile = blockIdx.x * 8 + wave;
  const int m0 = tile * 16;

  if (wave == 0) {                              // TDM: WoT -> LDS, once per workgroup
    tdm_load_1d_f16((unsigned)(uintptr_t)(void*)sWo, WoT, NPAD * KPAD);
    __builtin_amdgcn_s_wait_tensorcnt(0);
  }
  __syncthreads();

  __builtin_prefetch(upd + (size_t)(m0 + 128) * KPAD, 0, 0);

  v8f acc0 = {}, acc1 = {}, acc2 = {};
#pragma unroll
  for (int kk = 0; kk < 4; ++kk) {
    const v16h a  = load_frag_f16(upd + (size_t)m0 * KPAD, KPAD, kk * 32);
    const v16h b0 = load_frag_f16(sWo + 0 * 16 * KPAD, KPAD, kk * 32);
    const v16h b1 = load_frag_f16(sWo + 1 * 16 * KPAD, KPAD, kk * 32);
    const v16h b2 = load_frag_f16(sWo + 2 * 16 * KPAD, KPAD, kk * 32);
    acc0 = __builtin_amdgcn_wmma_f32_16x16x32_f16(false, a, false, b0, (short)0, acc0, false, false);
    acc1 = __builtin_amdgcn_wmma_f32_16x16x32_f16(false, a, false, b1, (short)0, acc1, false, false);
    acc2 = __builtin_amdgcn_wmma_f32_16x16x32_f16(false, a, false, b2, (short)0, acc2, false, false);
  }

  const int n  = lane & 15;
  const int mb = (lane >> 4) * 8;
  const v8f accs[3] = {acc0, acc1, acc2};
#pragma unroll
  for (int nt = 0; nt < 3; ++nt) {
    const int col = nt * 16 + n;
    if (col >= O_) continue;
    const float bias = bo[col];
#pragma unroll
    for (int r = 0; r < 8; ++r) {
      const int m = m0 + mb + r;
      out[(size_t)m * O_ + col] = tanhf(accs[nt][r] + bias);
    }
  }
}

extern "C" void kernel_launch(void* const* d_in, const int* in_sizes, int n_in,
                              void* d_out, int out_size, void* d_ws, size_t ws_size,
                              hipStream_t stream) {
  const float* x     = (const float*)d_in[0];
  const float* Ws    = (const float*)d_in[1];
  const float* bs    = (const float*)d_in[2];
  const float* Wf    = (const float*)d_in[3];
  const float* bf    = (const float*)d_in[4];
  const float* Wo    = (const float*)d_in[5];
  const float* bo_in = (const float*)d_in[6];
  float* out = (float*)d_out;
  char* ws = (char*)d_ws;

  float*    coords = (float*)(ws + WS_COORDS);
  float*    feats  = (float*)(ws + WS_FEATS);
  _Float16* upd    = (_Float16*)(ws + WS_UPD);
  _Float16* WcT    = (_Float16*)(ws + WS_WCT);
  float*    bcat   = (float*)(ws + WS_BCAT);
  _Float16* WoT    = (_Float16*)(ws + WS_WOT);
  float*    bo     = (float*)(ws + WS_BO);

  k_prep<<<33, 256, 0, stream>>>(Ws, bs, Wf, bf, Wo, bo_in, WcT, bcat, WoT, bo);
  k_gemm1<<<256, 256, 0, stream>>>(x, WcT, bcat, coords, feats, upd);
  k_knn<<<4096, 256, 0, stream>>>(coords, feats, upd);
  k_gemm2<<<256, 256, 0, stream>>>(upd, WoT, bo, out);
}
